// RecursiveAssociationNeuralNetworks_81243601371201
// MI455X (gfx1250) — matrix operand order repaired and checked by reading, not compiled, not run on hardware
//
#include <hip/hip_runtime.h>

typedef __attribute__((ext_vector_type(16))) _Float16 v16h;
typedef __attribute__((ext_vector_type(8)))  float    v8f;

#define NEGV (-1e9f)

struct __align__(16) WaveScratch {
  float ch[16 * 68];   // child_hiddens tile, row stride 68 floats (17 float4) -> bank-friendly
  float ac[16 * 20];   // A_c tile, row stride 20 floats (5 float4)
  float sd[32];        // src[0:16], dst[16:32]
  float pool[64];      // pooled readout
  float nf[40];        // node features
};

__global__ __launch_bounds__(256)
void rann_kernel(const float* __restrict__ nf_g,
                 const float* __restrict__ ch_g,
                 const float* __restrict__ ac_g,
                 const int*   __restrict__ cc_g,
                 const float* __restrict__ W_g,
                 const float* __restrict__ asrc_g,
                 const float* __restrict__ adst_g,
                 const float* __restrict__ Wx_g,
                 const float* __restrict__ Whg_g,
                 const float* __restrict__ b_g,
                 float* __restrict__ out_g,
                 int N)
{
  __shared__ float    sWx[40 * 192];
  __shared__ float    sWhg[64 * 192];
  __shared__ float    sB[192];
  __shared__ float    sAs[64];
  __shared__ float    sAd[64];
  __shared__ _Float16 sW[64 * 64];
  __shared__ WaveScratch sw_all[8];

  const int tid = threadIdx.x;

  // ---- block-level staging of shared weights (once) ----
  for (int i = tid; i < 40 * 192; i += 256) sWx[i]  = Wx_g[i];
  for (int i = tid; i < 64 * 192; i += 256) sWhg[i] = Whg_g[i];
  if (tid < 192) sB[tid] = b_g[tid];
  if (tid < 64)  { sAs[tid] = asrc_g[tid]; sAd[tid] = adst_g[tid]; }
  for (int i = tid; i < 64 * 64; i += 256) sW[i] = (_Float16)W_g[i];
  __syncthreads();

  const int lane = tid & 31;
  const int wv   = tid >> 5;
  const int g    = lane >> 4;   // half-wave group
  const int lm   = lane & 15;
  WaveScratch& sw = sw_all[wv];

  // ---- B-fragments of W (K x N = 64 x 64), kept in VGPRs for all nodes ----
  // B layout: element j of v16h, half g -> K = j + 16*g, N = 16*t + lm
  v16h WB[2][4];
  #pragma unroll
  for (int q = 0; q < 2; ++q)
    #pragma unroll
    for (int t = 0; t < 4; ++t)
      #pragma unroll
      for (int j = 0; j < 16; ++j)
        WB[q][t][j] = sW[(32 * q + 16 * g + j) * 64 + (16 * t + lm)];

  const int waveId = (blockIdx.x * blockDim.x + tid) >> 5;
  const int nWaves = (gridDim.x * blockDim.x) >> 5;

  for (int n = waveId; n < N; n += nWaves) {
    // prefetch next node's child tile (4KB) into cache
    {
      int n2 = n + nWaves;
      if (n2 < N) __builtin_prefetch(ch_g + (size_t)n2 * 1024 + lane * 32, 0, 0);
    }

    // ---- stage per-node data with coalesced b128 loads ----
    const float4* c4  = (const float4*)(ch_g + (size_t)n * 1024);
    float4*       chL = (float4*)sw.ch;
    #pragma unroll
    for (int i = 0; i < 8; ++i) {
      int s = i * 32 + lane;
      chL[(s >> 4) * 17 + (s & 15)] = c4[s];
    }
    const float4* a4  = (const float4*)(ac_g + (size_t)n * 256);
    float4*       acL = (float4*)sw.ac;
    #pragma unroll
    for (int i = 0; i < 2; ++i) {
      int s = i * 32 + lane;
      acL[(s >> 2) * 5 + (s & 3)] = a4[s];
    }
    sw.nf[lane] = nf_g[(size_t)n * 40 + lane];
    if (lane < 8) sw.nf[32 + lane] = nf_g[(size_t)n * 40 + 32 + lane];
    const int cnt = cc_g[n];

    // ---- A-fragments of child_hiddens (16 x 64, two K-chunks of 32) ----
    // A layout: element j, half g -> K = j + 16*g ; row M = lm
    v16h AF0, AF1;
    #pragma unroll
    for (int j = 0; j < 16; ++j) {
      int kb = j + 16 * g;
      AF0[j] = (_Float16)sw.ch[lm * 68 + kb];
      AF1[j] = (_Float16)sw.ch[lm * 68 + 32 + kb];
    }

    // ---- Wh = ch @ W : 8 WMMAs, f32 accumulate ----
    v8f wh[4];
    #pragma unroll
    for (int t = 0; t < 4; ++t) {
      v8f z = {};
      z = __builtin_amdgcn_wmma_f32_16x16x32_f16(false, AF0, false, WB[0][t], (short)0, z, false, false);
      z = __builtin_amdgcn_wmma_f32_16x16x32_f16(false, AF1, false, WB[1][t], (short)0, z, false, false);
      wh[t] = z;
    }

    // ---- src/dst = Wh . a_src / a_dst (D layout: lane holds col 16t+lm, rows r+8g) ----
    float ps[8] = {0,0,0,0,0,0,0,0}, pd[8] = {0,0,0,0,0,0,0,0};
    #pragma unroll
    for (int t = 0; t < 4; ++t) {
      float as_ = sAs[16 * t + lm], ad_ = sAd[16 * t + lm];
      #pragma unroll
      for (int r = 0; r < 8; ++r) { ps[r] += wh[t][r] * as_; pd[r] += wh[t][r] * ad_; }
    }
    #pragma unroll
    for (int m = 1; m <= 8; m <<= 1)
      #pragma unroll
      for (int r = 0; r < 8; ++r) {
        ps[r] += __shfl_xor(ps[r], m, 32);
        pd[r] += __shfl_xor(pd[r], m, 32);
      }
    if (lm == 0)
      #pragma unroll
      for (int r = 0; r < 8; ++r) {
        sw.sd[8 * g + r]      = ps[r];
        sw.sd[16 + 8 * g + r] = pd[r];
      }

    // ---- masked softmax over row i=lm; this lane owns cols jj = 8g..8g+7 ----
    float si = sw.sd[lm];
    float ev[8];
    #pragma unroll
    for (int k = 0; k < 8; ++k) {
      int jj = 8 * g + k;
      float x = si + sw.sd[16 + jj];
      x = (x > 0.0f) ? x : 0.2f * x;                       // leaky_relu(0.2)
      bool mk = (sw.ac[lm * 20 + jj] > 0.5f) && (jj < cnt) && (lm < cnt);
      ev[k] = mk ? x : NEGV;
    }
    float m8 = ev[0];
    #pragma unroll
    for (int k = 1; k < 8; ++k) m8 = fmaxf(m8, ev[k]);
    float M = fmaxf(m8, __shfl_xor(m8, 16, 32));
    float p[8], s8 = 0.0f;
    #pragma unroll
    for (int k = 0; k < 8; ++k) { p[k] = __expf(ev[k] - M); s8 += p[k]; }
    float S   = s8 + __shfl_xor(s8, 16, 32);
    float inv = 1.0f / S;

    // attn packed straight into A-fragment layout (K 16..31 = zero padding)
    v16h AT = {};
    #pragma unroll
    for (int k = 0; k < 8; ++k) AT[k] = (_Float16)(p[k] * inv);

    // ---- h_msg = attn @ Wh : build B'-fragments directly from this lane's own wh regs ----
    v8f hm[4];
    #pragma unroll
    for (int t = 0; t < 4; ++t) {
      v16h BP = {};
      #pragma unroll
      for (int r = 0; r < 8; ++r) {
        float ov = __shfl_xor(wh[t][r], 16, 32);           // rows 8..15 from partner lane
        BP[r]     = (_Float16)wh[t][r];
        BP[r + 8] = (_Float16)0.0f; // K>=16 half zeroed
      }
      v8f z = {};
      hm[t] = __builtin_amdgcn_wmma_f32_16x16x32_f16(false, AT, false, BP, (short)0, z, false, false);
    }

    // ---- ELU + masked max-pool over valid children ----
    #pragma unroll
    for (int t = 0; t < 4; ++t) {
      float pm = NEGV;
      #pragma unroll
      for (int r = 0; r < 8; ++r) {
        float x = hm[t][r];
        x = (x > 0.0f) ? x : (__expf(x) - 1.0f);           // elu
        int row = r + 8 * g;
        pm = (row < cnt) ? fmaxf(pm, x) : pm;
      }
      float po = fmaxf(pm, __shfl_xor(pm, 16, 32));
      if (g == 0) sw.pool[16 * t + lm] = po;
    }

    // ---- GRU update: each lane handles h = lane and h = lane+32 ----
    #pragma unroll
    for (int u = 0; u < 2; ++u) {
      int h = lane + 32 * u;
      float az = sB[h], ar = sB[64 + h], gxn = sB[128 + h];
      #pragma unroll 4
      for (int k = 0; k < 40; ++k) {
        float f = sw.nf[k];
        az  += f * sWx[k * 192 + h];
        ar  += f * sWx[k * 192 + 64 + h];
        gxn += f * sWx[k * 192 + 128 + h];
      }
      float hz = 0.0f, hr = 0.0f, hn = 0.0f;
      #pragma unroll 4
      for (int k = 0; k < 64; ++k) {
        float f = sw.pool[k];
        hz += f * sWhg[k * 192 + h];
        hr += f * sWhg[k * 192 + 64 + h];
        hn += f * sWhg[k * 192 + 128 + h];
      }
      float z  = 1.0f / (1.0f + __expf(-(az + hz)));
      float r  = 1.0f / (1.0f + __expf(-(ar + hr)));
      float nn = tanhf(gxn + r * hn);
      float pl = sw.pool[h];
      out_g[(size_t)n * 64 + h] = (1.0f - z) * nn + z * pl;
    }
  }
}

extern "C" void kernel_launch(void* const* d_in, const int* in_sizes, int n_in,
                              void* d_out, int out_size, void* d_ws, size_t ws_size,
                              hipStream_t stream) {
  const float* nf  = (const float*)d_in[0];
  const float* ch  = (const float*)d_in[1];
  const float* ac  = (const float*)d_in[2];
  const int*   cc  = (const int*)d_in[3];
  const float* W   = (const float*)d_in[4];
  const float* as_ = (const float*)d_in[5];
  const float* ad_ = (const float*)d_in[6];
  const float* Wx  = (const float*)d_in[7];
  const float* Whg = (const float*)d_in[8];
  const float* b   = (const float*)d_in[9];
  float* out = (float*)d_out;

  int N = in_sizes[0] / 40;          // node_features is [N,1,40]
  int blocks = 2048;                 // 16384 persistent waves, ~12 nodes each
  rann_kernel<<<blocks, 256, 0, stream>>>(nf, ch, ac, cc, W, as_, ad_, Wx, Whg, b, out, N);
}
